// FourierHead_35639638622763
// MI455X (gfx1250) — compile-verified, matrix operand order checked by reading, not run on hardware
//
#include <hip/hip_runtime.h>
#include <hip/hip_bf16.h>
#include <math.h>

typedef __attribute__((ext_vector_type(16))) __bf16 v16bf;
typedef __attribute__((ext_vector_type(8)))  float  v8f;

#define DIMK   1024
#define NCOL   514      // 2*(m+1)
#define NPADW  576      // padded row count of bf16 W (multiple of BN)
#define NPAD   528      // padded row stride of staged P
#define BM     128
#define BN     64
#define KB     64
#define NKB    (DIMK / KB)   // 16
#define LDSX_S 72       // 64 bf16 + 8 pad  (16B pad -> conflict-free b128 frag reads)
#define LDSW_S 72

__device__ __forceinline__ unsigned short f2bf(float f) {
  unsigned u = __float_as_uint(f);
  u += 0x7FFFu + ((u >> 16) & 1u);   // round-to-nearest-even
  return (unsigned short)(u >> 16);
}
__device__ __forceinline__ unsigned pk_bf16(float lo, float hi) {
  return (unsigned)f2bf(lo) | ((unsigned)f2bf(hi) << 16);
}

// CDNA5 async global->LDS DMA copy (ASYNCcnt-tracked), GV addressing mode.
__device__ __forceinline__ void async_ld_b128(unsigned ldsOff, const void* gaddr) {
  asm volatile("global_load_async_to_lds_b128 %0, %1, off"
               :: "v"(ldsOff), "v"((unsigned long long)(uintptr_t)gaddr)
               : "memory");
}
__device__ __forceinline__ void wait_async0() {
  asm volatile("s_wait_asynccnt 0x0" ::: "memory");
}

union FragU { v16bf v; uint4 q[2]; };

// ---- one-shot f32 -> bf16 conversion (memory-bound; off the WMMA critical path)
__global__ __launch_bounds__(256)
void cvt_x_bf16(const float* __restrict__ src, unsigned* __restrict__ dst)
{
  int gid = blockIdx.x * 256 + threadIdx.x;        // one id per 8 elements
  const float4* s = (const float4*)src + (size_t)gid * 2;
  float4 a = s[0], b = s[1];
  ((uint4*)dst)[gid] = make_uint4(pk_bf16(a.x, a.y), pk_bf16(a.z, a.w),
                                  pk_bf16(b.x, b.y), pk_bf16(b.z, b.w));
}

// W: convert 514 rows, zero-pad rows 514..575 so GEMM needs no guards
__global__ __launch_bounds__(256)
void cvt_w_bf16(const float* __restrict__ W, unsigned* __restrict__ dst)
{
  int gid = blockIdx.x * 256 + threadIdx.x;        // 576*1024/8 = 73728 ids
  int row = gid >> 7;                              // 128 ids per 1024-wide row
  uint4 o;
  if (row < NCOL) {
    const float4* s = (const float4*)W + (size_t)gid * 2;
    float4 a = s[0], b = s[1];
    o = make_uint4(pk_bf16(a.x, a.y), pk_bf16(a.z, a.w),
                   pk_bf16(b.x, b.y), pk_bf16(b.z, b.w));
  } else {
    o = make_uint4(0u, 0u, 0u, 0u);
  }
  ((uint4*)dst)[gid] = o;
}

// ---- bf16 WMMA GEMM: P = Xb @ Wb^T + bias, async-DMA fed, LDS double-buffered
__global__ __launch_bounds__(256)
void fourier_gemm(const unsigned short* __restrict__ Xb,
                  const unsigned short* __restrict__ Wb,
                  const float* __restrict__ bias, float* __restrict__ P)
{
  __shared__ unsigned short lx[2][BM * LDSX_S];   // X tiles (bf16)
  __shared__ unsigned short lw[2][BN * LDSW_S];   // W tiles (bf16)

  const int t     = threadIdx.x;
  const int lane  = t & 31;
  const int wave  = t >> 5;
  const int waveM = wave >> 1;       // 0..3
  const int waveN = wave & 1;        // 0..1
  const int blkN  = blockIdx.x;      // 0..8
  const int blkM  = blockIdx.y;      // 0..255

  const size_t rowM0 = (size_t)blkM * BM;
  const int    colN0 = blkN * BN;

  // issue per-thread async copies for K-block kb into LDS buffer `buf`
  auto issue = [&](int kb, int buf) {
    const int kbase = kb * KB;
    #pragma unroll
    for (int i = 0; i < 4; ++i) {            // X: 128x64 bf16 = 1024 b128 chunks
      int id = i * 256 + t;
      int r  = id >> 3;                      // 0..127
      int g  = id & 7;                       // 8-elem group in 64-wide row
      unsigned lo = (unsigned)(uintptr_t)&lx[buf][r * LDSX_S + g * 8];
      async_ld_b128(lo, Xb + (rowM0 + r) * DIMK + kbase + g * 8);
    }
    #pragma unroll
    for (int i = 0; i < 2; ++i) {            // W: 64x64 bf16 = 512 b128 chunks
      int id = i * 256 + t;
      int r  = id >> 3;                      // 0..63
      int g  = id & 7;
      unsigned lo = (unsigned)(uintptr_t)&lw[buf][r * LDSW_S + g * 8];
      async_ld_b128(lo, Wb + (size_t)(colN0 + r) * DIMK + kbase + g * 8);
    }
  };

  v8f zero = {};
  v8f acc[2][2];
  acc[0][0] = zero; acc[0][1] = zero; acc[1][0] = zero; acc[1][1] = zero;

  // ISA 16-bit A/B fragment addressing: lanes 0-15 hold K {0..7,16..23},
  // lanes 16-31 hold K {8..15,24..31}; row/col = lane & 15.
  const int rA   = lane & 15;
  const int kSel = (lane >> 4) << 3;       // 0 or 8

  // compute one K-block from LDS buffer `buf` (buf is a compile-time constant)
  auto compute = [&](const int buf) {
    #pragma unroll
    for (int kk = 0; kk < KB; kk += 32) {
      FragU a[2], bf[2];
      #pragma unroll
      for (int mt = 0; mt < 2; ++mt) {
        int row = waveM * 32 + mt * 16 + rA;
        const unsigned short* s = &lx[buf][row * LDSX_S + kk + kSel];
        a[mt].q[0] = *(const uint4*)(s);
        a[mt].q[1] = *(const uint4*)(s + 16);
      }
      #pragma unroll
      for (int nt = 0; nt < 2; ++nt) {
        int row = waveN * 32 + nt * 16 + rA;
        const unsigned short* s = &lw[buf][row * LDSW_S + kk + kSel];
        bf[nt].q[0] = *(const uint4*)(s);
        bf[nt].q[1] = *(const uint4*)(s + 16);
      }
      #pragma unroll
      for (int mt = 0; mt < 2; ++mt)
        #pragma unroll
        for (int nt = 0; nt < 2; ++nt)
          acc[mt][nt] = __builtin_amdgcn_wmma_f32_16x16x32_bf16(
              false, a[mt].v, false, bf[nt].v, (short)0, acc[mt][nt],
              false, false);
    }
  };

  issue(0, 0);
  // manual 2x unroll so the buffer index is a literal constant in each phase
  for (int kb2 = 0; kb2 < NKB; kb2 += 2) {
    // phase A: consume buf 0, prefetch into buf 1
    wait_async0();        // own async writes (into buf 0) complete
    __syncthreads();      // all waves' writes complete; all done reading buf 1
    if (kb2 + 1 < NKB) issue(kb2 + 1, 1);
    compute(0);

    // phase B: consume buf 1, prefetch into buf 0
    wait_async0();        // own async writes (into buf 1) complete
    __syncthreads();      // all waves' writes complete; all done reading buf 0
    if (kb2 + 2 < NKB) issue(kb2 + 2, 0);
    compute(1);
  }

  // Epilogue: D layout — VGPR i, lanes 0-15 -> M=i, lanes 16-31 -> M=i+8; N = lane&15.
  const int cN   = lane & 15;
  const int rOff = (lane >> 4) * 8;
  #pragma unroll
  for (int mt = 0; mt < 2; ++mt) {
    #pragma unroll
    for (int nt = 0; nt < 2; ++nt) {
      int col = colN0 + waveN * 32 + nt * 16 + cN;
      if (col < NCOL) {
        float bb = bias[col];
        size_t rowBase = rowM0 + waveM * 32 + mt * 16 + rOff;
        #pragma unroll
        for (int i = 0; i < 8; ++i)
          P[(rowBase + i) * NPAD + col] = acc[mt][nt][i] + bb;
      }
    }
  }
}

// pdf(b) = |sum_j a_j e^{i j pi t}|^2 / (2 * sum_j |a_j|^2)
__global__ __launch_bounds__(256)
void fourier_eval(const float* __restrict__ P, const float* __restrict__ tgt,
                  float* __restrict__ out)
{
  const int lane = threadIdx.x & 31;
  const int wave = threadIdx.x >> 5;
  const int b    = blockIdx.x * 8 + wave;

  const float t  = tgt[b];
  const float th = 3.14159265358979323846f * t;
  float c, s, c32, s32;
  sincosf((float)lane * th, &s, &c);     // angle for j = lane
  sincosf(32.0f * th, &s32, &c32);       // step by 32 lanes

  const float* row = P + (size_t)b * NPAD;
  float sumR = 0.f, sumI = 0.f, a0 = 0.f;
  #pragma unroll
  for (int it = 0; it < 9; ++it) {
    int j = it * 32 + lane;
    if (j < 257) {
      float ar = row[j];
      float ai = row[257 + j];
      sumR = fmaf(ar, c, fmaf(-ai, s, sumR));
      sumI = fmaf(ar, s, fmaf( ai, c, sumI));
      a0   = fmaf(ar, ar, fmaf(ai, ai, a0));
    }
    float cn = c * c32 - s * s32;
    s = s * c32 + c * s32;
    c = cn;
  }
  #pragma unroll
  for (int off = 16; off > 0; off >>= 1) {
    sumR += __shfl_xor(sumR, off, 32);
    sumI += __shfl_xor(sumI, off, 32);
    a0   += __shfl_xor(a0,   off, 32);
  }
  if (lane == 0)
    out[b] = (sumR * sumR + sumI * sumI) / (2.0f * a0);
}

extern "C" void kernel_launch(void* const* d_in, const int* in_sizes, int n_in,
                              void* d_out, int out_size, void* d_ws, size_t ws_size,
                              hipStream_t stream) {
  (void)in_sizes; (void)n_in; (void)out_size; (void)ws_size;
  const float* X    = (const float*)d_in[0];   // (32768, 1024) f32
  const float* tgt  = (const float*)d_in[1];   // (32768,)
  const float* W    = (const float*)d_in[2];   // (514, 1024) f32
  const float* bias = (const float*)d_in[3];   // (514,)

  // workspace layout (all 16B aligned)
  char* ws = (char*)d_ws;
  unsigned short* Xb = (unsigned short*)ws;                               // 67,108,864 B
  unsigned short* Wb = (unsigned short*)(ws + (size_t)32768 * DIMK * 2);  //  1,179,648 B
  float*          P  = (float*)(ws + (size_t)32768 * DIMK * 2
                                   + (size_t)NPADW * DIMK * 2);           // 69,206,016 B

  cvt_x_bf16<<<dim3(32768 * DIMK / 8 / 256, 1, 1), dim3(256, 1, 1), 0, stream>>>(X, (unsigned*)Xb);
  cvt_w_bf16<<<dim3(NPADW * DIMK / 8 / 256, 1, 1), dim3(256, 1, 1), 0, stream>>>(W, (unsigned*)Wb);

  dim3 g1((NCOL + BN - 1) / BN, 32768 / BM, 1);   // 9 x 256
  fourier_gemm<<<g1, dim3(256, 1, 1), 0, stream>>>(Xb, Wb, bias, P);
  fourier_eval<<<dim3(32768 / 8, 1, 1), dim3(256, 1, 1), 0, stream>>>(P, tgt, (float*)d_out);
}
